// PointBlock_51719996178710
// MI455X (gfx1250) — 8-bit matrix operand order repaired and checked by reading, compiled, not run on hardware
//
#include <hip/hip_runtime.h>
#include <hip/hip_bf16.h>

#define HW     3136
#define NC     256
#define NB     32
#define MTOT   100352.0

typedef __attribute__((ext_vector_type(8))) int v8i;

__global__ __launch_bounds__(256) void wprep_kernel(
    const float* __restrict__ w, float* __restrict__ scale,
    signed char* __restrict__ swb, double* __restrict__ gsum, double* __restrict__ gsq)
{
    __shared__ float red[256];
    const int o = blockIdx.x;
    const int t = threadIdx.x;
    const float wv = w[o * NC + t];
    red[t] = fabsf(wv);
    __syncthreads();
    for (int s = 128; s > 0; s >>= 1) {
        if (t < s) red[t] += red[t + s];
        __syncthreads();
    }
    if (t == 0) scale[o] = red[0] * (1.0f / 256.0f);
    swb[o * NC + t] = (wv >= 0.0f) ? (signed char)1 : (signed char)-1;
    if (blockIdx.x == 0) { gsum[t] = 0.0; gsq[t] = 0.0; }
}

__global__ __launch_bounds__(256) void signT_kernel(
    const float* __restrict__ x, const float* __restrict__ bias0,
    signed char* __restrict__ sbT)
{
    __shared__ signed char tile[64 * 68];
    const int n  = blockIdx.z;
    const int c0 = blockIdx.y * 64;
    const int p0 = blockIdx.x * 64;
    const int t  = threadIdx.x;

    const int cr = t >> 4;
    const int pc = (t & 15) << 2;
#pragma unroll
    for (int k = 0; k < 4; ++k) {
        const int c = cr + k * 16;
        const float4 v = *(const float4*)(x + (size_t)(n * NC + c0 + c) * HW + p0 + pc);
        const float b = bias0[c0 + c];
        char4 s;
        s.x = (v.x + b >= 0.0f) ? 1 : -1;
        s.y = (v.y + b >= 0.0f) ? 1 : -1;
        s.z = (v.z + b >= 0.0f) ? 1 : -1;
        s.w = (v.w + b >= 0.0f) ? 1 : -1;
        *(char4*)&tile[c * 68 + pc] = s;
    }
    __syncthreads();

    const int p  = t & 63;
    const int cc = (t >> 6) << 4;
    unsigned int wds[4];
#pragma unroll
    for (int j = 0; j < 4; ++j) {
        unsigned int acc = 0;
#pragma unroll
        for (int i = 0; i < 4; ++i) {
            const unsigned int byt = (unsigned char)tile[(cc + j * 4 + i) * 68 + p];
            acc |= byt << (8 * i);
        }
        wds[j] = acc;
    }
    uint4 outw; outw.x = wds[0]; outw.y = wds[1]; outw.z = wds[2]; outw.w = wds[3];
    *(uint4*)(sbT + (size_t)(n * HW + p0 + p) * NC + c0 + cc) = outw;
}

#define ASTR 272
template<bool STATS>
__global__ __launch_bounds__(256) void gemm_kernel(
    const signed char* __restrict__ sbT, const signed char* __restrict__ swb,
    const float* __restrict__ x, float* __restrict__ out,
    const float* __restrict__ a_bn, const float* __restrict__ b_bn,
    const float* __restrict__ bias1, const float* __restrict__ alpha,
    const float* __restrict__ bias2,
    double* __restrict__ gsum, double* __restrict__ gsq)
{
    __shared__ signed char As[128 * ASTR];
    __shared__ signed char Bs[64 * ASTR];
    __shared__ float fsum[128], fsq[128];
    __shared__ float pA[128], pB[128], p1[128], pal[128], p2[128];

    const int t     = threadIdx.x;
    const int n     = blockIdx.z;
    const int oBase = blockIdx.y * 128;
    const int pBase = blockIdx.x * 64;

    {
        const uint4* gA = (const uint4*)(swb + (size_t)oBase * NC);
#pragma unroll
        for (int i = 0; i < 8; ++i) {
            const int idx = t + i * 256;
            const int row = idx >> 4, seg = idx & 15;
            *(uint4*)&As[row * ASTR + seg * 16] = gA[row * 16 + seg];
        }
        const uint4* gB = (const uint4*)(sbT + ((size_t)n * HW + pBase) * NC);
#pragma unroll
        for (int i = 0; i < 4; ++i) {
            const int idx = t + i * 256;
            const int row = idx >> 4, seg = idx & 15;
            *(uint4*)&Bs[row * ASTR + seg * 16] = gB[row * 16 + seg];
        }
    }
    if (STATS) {
        if (t < 128) { fsum[t] = 0.0f; fsq[t] = 0.0f; }
    } else {
        if (t < 128) {
            const int o = oBase + t;
            pA[t] = a_bn[o]; pB[t] = b_bn[o];
            p1[t] = bias1[o]; pal[t] = alpha[o]; p2[t] = bias2[o];
        }
    }
    __syncthreads();

    const int  wave = t >> 5;
    const int  lane = t & 31;
    const int  lr   = lane & 15;
    const bool hi   = lane >= 16;
    const int  wo   = (wave & 3) * 32;
    const int  wp   = (wave >> 2) * 32;

    v8i acc[2][2] = {};
#pragma unroll
    for (int kc = 0; kc < 256; kc += 64) {
        v8i a[2], b[2];
#pragma unroll
        for (int mt = 0; mt < 2; ++mt) {
            const signed char* ap = &As[(wo + mt * 16 + lr) * ASTR + kc + (hi ? 8 : 0)];
            const uint2 q0 = *(const uint2*)(ap +  0);
            const uint2 q1 = *(const uint2*)(ap + 16);
            const uint2 q2 = *(const uint2*)(ap + 32);
            const uint2 q3 = *(const uint2*)(ap + 48);
            v8i av = {(int)q0.x, (int)q0.y, (int)q1.x, (int)q1.y,
                      (int)q2.x, (int)q2.y, (int)q3.x, (int)q3.y};
            a[mt] = av;
        }
#pragma unroll
        for (int nt = 0; nt < 2; ++nt) {
            const signed char* bp = &Bs[(wp + nt * 16 + lr) * ASTR + kc + (hi ? 8 : 0)];
            const uint2 r0 = *(const uint2*)(bp +  0);
            const uint2 r1 = *(const uint2*)(bp + 16);
            const uint2 r2 = *(const uint2*)(bp + 32);
            const uint2 r3 = *(const uint2*)(bp + 48);
            v8i bv = {(int)r0.x, (int)r0.y, (int)r1.x, (int)r1.y,
                      (int)r2.x, (int)r2.y, (int)r3.x, (int)r3.y};
            b[nt] = bv;
        }
#pragma unroll
        for (int mt = 0; mt < 2; ++mt)
#pragma unroll
            for (int nt = 0; nt < 2; ++nt)
                acc[mt][nt] = __builtin_amdgcn_wmma_i32_16x16x64_iu8(
                    true, a[mt], true, b[nt], acc[mt][nt], false, false);
    }

    if (STATS) {
#pragma unroll
        for (int mt = 0; mt < 2; ++mt) {
#pragma unroll
            for (int v = 0; v < 8; ++v) {
                float s = 0.0f, s2 = 0.0f;
#pragma unroll
                for (int nt = 0; nt < 2; ++nt) {
                    const float y = (float)acc[mt][nt][v];
                    s += y; s2 += y * y;
                }
#pragma unroll
                for (int m = 8; m >= 1; m >>= 1) {
                    s  += __shfl_xor(s,  m, 32);
                    s2 += __shfl_xor(s2, m, 32);
                }
                if (lr == 0) {
                    const int ol = wo + mt * 16 + v + (hi ? 8 : 0);
                    atomicAdd(&fsum[ol], s);
                    atomicAdd(&fsq[ol], s2);
                }
            }
        }
        __syncthreads();
        if (t < 128) {
            atomicAdd(&gsum[oBase + t], (double)fsum[t]);
            atomicAdd(&gsq[oBase + t],  (double)fsq[t]);
        }
    } else {
#pragma unroll
        for (int mt = 0; mt < 2; ++mt)
#pragma unroll
            for (int nt = 0; nt < 2; ++nt)
#pragma unroll
                for (int v = 0; v < 8; ++v) {
                    const int ol = wo + mt * 16 + v + (hi ? 8 : 0);
                    const int p  = pBase + wp + nt * 16 + lr;
                    const size_t gi = (size_t)(n * NC + oBase + ol) * HW + p;
                    float y = pA[ol] * (float)acc[mt][nt][v] + pB[ol] + x[gi] + p1[ol];
                    y = (y > 0.0f) ? y : pal[ol] * y;
                    out[gi] = y + p2[ol];
                }
    }
}

__global__ __launch_bounds__(256) void bnprep_kernel(
    const double* __restrict__ gsum, const double* __restrict__ gsq,
    const float* __restrict__ scale, const float* __restrict__ gamma,
    const float* __restrict__ beta, float* __restrict__ a_bn, float* __restrict__ b_bn)
{
    const int o = threadIdx.x;
    const double mui  = gsum[o] / MTOT;
    const double vari = gsq[o] / MTOT - mui * mui;
    const float sc  = scale[o];
    const float inv = rsqrtf((float)vari * sc * sc + 1e-5f);
    const float g   = gamma[o];
    a_bn[o] = g * inv * sc;
    b_bn[o] = beta[o] - g * inv * sc * (float)mui;
}

extern "C" void kernel_launch(void* const* d_in, const int* in_sizes, int n_in,
                              void* d_out, int out_size, void* d_ws, size_t ws_size,
                              hipStream_t stream) {
    const float* x     = (const float*)d_in[0];
    const float* bias0 = (const float*)d_in[1];
    const float* w     = (const float*)d_in[2];
    const float* gamma = (const float*)d_in[3];
    const float* beta  = (const float*)d_in[4];
    const float* bias1 = (const float*)d_in[5];
    const float* alpha = (const float*)d_in[6];
    const float* bias2 = (const float*)d_in[7];
    float* out = (float*)d_out;

    char* ws = (char*)d_ws;
    size_t off = 0;
    signed char* sbT = (signed char*)(ws + off); off += (size_t)NB * HW * NC;
    signed char* swb = (signed char*)(ws + off); off += (size_t)NC * NC;
    float* scale = (float*)(ws + off); off += 1024;
    float* a_bn  = (float*)(ws + off); off += 1024;
    float* b_bn  = (float*)(ws + off); off += 1024;
    double* gsum = (double*)(ws + off); off += 2048;
    double* gsq  = (double*)(ws + off); off += 2048;

    wprep_kernel<<<256, 256, 0, stream>>>(w, scale, swb, gsum, gsq);
    signT_kernel<<<dim3(49, 4, 32), 256, 0, stream>>>(x, bias0, sbT);
    gemm_kernel<true><<<dim3(49, 2, 32), 256, 0, stream>>>(
        sbT, swb, x, out, a_bn, b_bn, bias1, alpha, bias2, gsum, gsq);
    bnprep_kernel<<<1, 256, 0, stream>>>(gsum, gsq, scale, gamma, beta, a_bn, b_bn);
    gemm_kernel<false><<<dim3(49, 2, 32), 256, 0, stream>>>(
        sbT, swb, x, out, a_bn, b_bn, bias1, alpha, bias2, gsum, gsq);
}
